// TextLSTM_85237920957100
// MI455X (gfx1250) — matrix operand order repaired and checked by reading, not compiled, not run on hardware
//
#include <hip/hip_runtime.h>
#include <hip/hip_bf16.h>
#include <stdint.h>

// ---------------------------------------------------------------------------
// TextLSTM forward on gfx1250 (MI455X): bf16 WMMA for input projections and
// recurrence, streaming FMA for the memory-bound vocab projection.
// Recurrence: persistent kernel, double-buffered packed H (1 grid barrier per
// step), software-pipelined K loop, prefetch of next step's preactivations.
// ---------------------------------------------------------------------------

typedef __attribute__((ext_vector_type(16))) __bf16 v16bf;
typedef __attribute__((ext_vector_type(8)))  float  v8f;

#define N_CLASS 50257
#define EMB     256
#define HID     512
#define BATCH   64
#define SEQ     256
#define NG      2048            // 4 * HID (i,f,o,c concatenated)
#define NT_ALL  128             // NG / 16 column tiles
#define KT_X    8               // EMB / 32
#define KT_H    16              // HID / 32
#define R_BLOCKS 16             // persistent blocks in recurrence kernel

// A-operand (16x32 bf16) per-lane K packing (ISA 7.12.2):
//   lane l holds row m = l&15; element e -> k = (e>=8?16:0) + ((l>>4)<<3) + (e&7)
__device__ __forceinline__ int a_klocal(int lane, int e) {
    return ((e & 8) << 1) | (((lane >> 4) & 1) << 3) | (e & 7);
}
// B-operand (32x16 bf16): lane l holds col n = l&15;
//   element e -> k = (e>=8?16:0) + ((l>>4)<<3) + (e&7)   (same K order as A)
__device__ __forceinline__ int b_klocal(int lane, int e) {
    return ((e & 8) << 1) | (((lane >> 4) & 1) << 3) | (e & 7);
}

__device__ __forceinline__ float sigm(float x) {
    return 1.0f / (1.0f + __expf(-x));
}

// Grid-wide barrier: monotonic counter, one arrival per block.
__device__ __forceinline__ void grid_sync(unsigned* bar, unsigned target) {
    __threadfence();
    __syncthreads();
    if (threadIdx.x == 0) {
        atomicAdd(bar, 1u);
        while (__hip_atomic_load(bar, __ATOMIC_RELAXED, __HIP_MEMORY_SCOPE_AGENT) < target) {
            __builtin_amdgcn_s_sleep(2);
        }
    }
    __syncthreads();
    __threadfence();
}

// ---------------------------------------------------------------------------
// Kernel P: pack W_x* [256,512]x4 and W_h* [512,512]x4 (fp32) into bf16 tiles
// laid out exactly as WMMA B-operand registers:
//   packed[(kt*128 + nt)*32 + lane][16]  -> one 32B contiguous chunk per lane.
// Column space n = g*512 + col, g in {i,f,o,c}.
// ---------------------------------------------------------------------------
__global__ __launch_bounds__(256) void pack_weights(
    const float* __restrict__ Wxi, const float* __restrict__ Wxf,
    const float* __restrict__ Wxo, const float* __restrict__ Wxc,
    const float* __restrict__ Whi, const float* __restrict__ Whf,
    const float* __restrict__ Who, const float* __restrict__ Whc,
    __bf16* __restrict__ pWx, __bf16* __restrict__ pWh)
{
    int tid = blockIdx.x * blockDim.x + threadIdx.x;
    const int TX = KT_X * NT_ALL * 32;   // 32768 lane-slots
    const int TH = KT_H * NT_ALL * 32;   // 65536 lane-slots
    if (tid < TX) {
        int lane = tid & 31, nt = (tid >> 5) & (NT_ALL - 1), kt = tid >> 12;
        const float* W[4] = {Wxi, Wxf, Wxo, Wxc};
        int n = nt * 16 + (lane & 15);
        int g = n >> 9, c = n & (HID - 1);
        __bf16* dst = pWx + (((size_t)kt * NT_ALL + nt) * 32 + lane) * 16;
        const float* src = W[g];
        #pragma unroll
        for (int e = 0; e < 16; ++e) {
            int k = kt * 32 + b_klocal(lane, e);
            dst[e] = (__bf16)src[(size_t)k * HID + c];
        }
    } else if (tid < TX + TH) {
        int id = tid - TX;
        int lane = id & 31, nt = (id >> 5) & (NT_ALL - 1), kt = id >> 12;
        const float* W[4] = {Whi, Whf, Who, Whc};
        int n = nt * 16 + (lane & 15);
        int g = n >> 9, c = n & (HID - 1);
        __bf16* dst = pWh + (((size_t)kt * NT_ALL + nt) * 32 + lane) * 16;
        const float* src = W[g];
        #pragma unroll
        for (int e = 0; e < 16; ++e) {
            int k = kt * 32 + b_klocal(lane, e);
            dst[e] = (__bf16)src[(size_t)k * HID + c];
        }
    }
}

// ---------------------------------------------------------------------------
// Kernel I: copy C into ws (inputs must not be mutated), pack H into the
// bf16 A-operand shadow buffer pH0[(kt*4 + mt)*32 + lane][16], zero barrier.
// ---------------------------------------------------------------------------
__global__ __launch_bounds__(256) void init_state(
    const float* __restrict__ Hin, const float* __restrict__ Cin,
    __bf16* __restrict__ pH0, float* __restrict__ Cws, unsigned* __restrict__ bar)
{
    int tid = blockIdx.x * blockDim.x + threadIdx.x;
    if (tid == 0) bar[0] = 0u;
    if (tid < BATCH * HID) {
        Cws[tid] = Cin[tid];
        int b = tid >> 9, h = tid & (HID - 1);
        int mt = b >> 4, m = b & 15;
        int kt = h >> 5, kl = h & 31;
        int e    = ((kl & 16) ? 8 : 0) + (kl & 7);     // inverse of a_klocal
        int lane = (((kl >> 3) & 1) << 4) + m;
        pH0[(((size_t)kt * 4 + mt) * 32 + lane) * 16 + e] = (__bf16)Hin[tid];
    }
}

// ---------------------------------------------------------------------------
// Kernel B: gate pre-activations  xproj[t][b][n] = E[X] @ Wx_g + bias_g
// Rows r = t*64 + b (16384 rows), cols n in [0,2048). One wave handles one
// 16-row tile x four 16-col tiles; A gathered from E fp32 -> bf16 per kt.
// ---------------------------------------------------------------------------
__global__ __launch_bounds__(256) void xproj_kernel(
    const int* __restrict__ X, const float* __restrict__ E,
    const __bf16* __restrict__ pWx,
    const float* __restrict__ bi, const float* __restrict__ bfv,
    const float* __restrict__ bo, const float* __restrict__ bc,
    float* __restrict__ xproj)
{
    const int lane = threadIdx.x & 31;
    const int wv   = threadIdx.x >> 5;
    const int mt   = blockIdx.x >> 2;                 // 0..1023
    const int nq   = blockIdx.x & 3;
    const int ntb  = nq * 32 + wv * 4;                // first of 4 n-tiles

    // token for this lane's row (lanes 16-31 carry the same rows, other K half)
    const int m = lane & 15;
    const int r = mt * 16 + m;
    const int t = r >> 6, b = r & 63;
    const int tok = X[b * SEQ + t];

    v8f acc0 = 0, acc1 = 0, acc2 = 0, acc3 = 0;

    for (int kt = 0; kt < KT_X; ++kt) {
        v16bf a;
        #pragma unroll
        for (int e = 0; e < 16; ++e) {
            int k = kt * 32 + a_klocal(lane, e);
            a[e] = (__bf16)E[(size_t)tok * EMB + k];
        }
        const __bf16* wb = pWx + (((size_t)kt * NT_ALL + ntb) * 32 + lane) * 16;
        v16bf b0 = *(const v16bf*)(wb);
        v16bf b1 = *(const v16bf*)(wb + 32 * 16);
        v16bf b2 = *(const v16bf*)(wb + 2 * 32 * 16);
        v16bf b3 = *(const v16bf*)(wb + 3 * 32 * 16);
        acc0 = __builtin_amdgcn_wmma_f32_16x16x32_bf16(false, a, false, b0, (short)0, acc0, false, false);
        acc1 = __builtin_amdgcn_wmma_f32_16x16x32_bf16(false, a, false, b1, (short)0, acc1, false, false);
        acc2 = __builtin_amdgcn_wmma_f32_16x16x32_bf16(false, a, false, b2, (short)0, acc2, false, false);
        acc3 = __builtin_amdgcn_wmma_f32_16x16x32_bf16(false, a, false, b3, (short)0, acc3, false, false);
    }

    // store with bias. D layout: lane l, reg v -> row m=(l>>4)*8+v, col n=l&15.
    const int nl = lane & 15;
    const int mb = mt * 16 + ((lane >> 4) << 3);
    v8f accs[4] = {acc0, acc1, acc2, acc3};
    #pragma unroll
    for (int j = 0; j < 4; ++j) {
        int n = (ntb + j) * 16 + nl;
        int g = n >> 9, c = n & (HID - 1);
        const float* bp = (g == 0) ? bi : (g == 1) ? bfv : (g == 2) ? bo : bc;
        float bias = bp[c];
        #pragma unroll
        for (int v = 0; v < 8; ++v) {
            xproj[(size_t)(mb + v) * NG + n] = accs[j][v] + bias;
        }
    }
}

// ---------------------------------------------------------------------------
// Kernel R: persistent recurrence. 16 blocks x 8 waves = 128 waves; wave w
// owns D-tile (mt = w>>5, ht = w&31) and computes all four gates for it, so
// the C/H pointwise update is wave-local. The packed-A H shadow is double
// buffered (read t&1, write the other) so only ONE grid barrier per step is
// needed. K loop is software-pipelined: loads for kt+1 overlap WMMAs for kt.
// ---------------------------------------------------------------------------
__global__ __launch_bounds__(256) void lstm_recurrence(
    const float* __restrict__ xproj, const __bf16* __restrict__ pWh,
    __bf16* __restrict__ pH0, __bf16* __restrict__ pH1,
    float* __restrict__ Cws, float* __restrict__ Hf32, unsigned* __restrict__ bar)
{
    const int lane = threadIdx.x & 31;
    const int wave = blockIdx.x * 8 + (threadIdx.x >> 5);   // 0..127
    const int mt   = wave >> 5;                             // 0..3
    const int ht   = wave & 31;                             // 0..31
    const int nl   = lane & 15;
    const int mb   = mt * 16 + ((lane >> 4) << 3);          // row base for regs

    // scatter constants: where column h_col lands in the packed-A H buffer
    const int h_col = ht * 16 + nl;
    const int s_kt  = h_col >> 5;
    const int kl    = h_col & 31;
    const int s_e   = ((kl & 16) ? 8 : 0) + (kl & 7);
    const int s_khb = ((kl >> 3) & 1) << 4;

    // per-wave constant offsets into packed buffers
    const size_t aoff = ((size_t)mt * 32 + lane) * 16;        // + kt*2048
    size_t boff[4];
    #pragma unroll
    for (int g = 0; g < 4; ++g)
        boff[g] = (((size_t)(g * 32 + ht)) * 32 + lane) * 16; // + kt*65536

    for (int t = 0; t < SEQ; ++t) {
        const __bf16* pHr = (t & 1) ? pH1 : pH0;
        __bf16*       pHw = (t & 1) ? pH0 : pH1;
        const float*  xp  = xproj + (size_t)t * BATCH * NG;

        v8f acc[4];
        #pragma unroll
        for (int g = 0; g < 4; ++g) {
            #pragma unroll
            for (int v = 0; v < 8; ++v)
                acc[g][v] = xp[(size_t)(mb + v) * NG + g * HID + h_col];
        }

        // prefetch next step's preactivation lines while WMMAs run
        if (t + 1 < SEQ) {
            const float* xpn = xp + (size_t)BATCH * NG;
            #pragma unroll
            for (int g = 0; g < 4; ++g)
                #pragma unroll
                for (int v = 0; v < 8; ++v)
                    __builtin_prefetch(&xpn[(size_t)(mb + v) * NG + g * HID + h_col], 0, 3);
        }

        // software-pipelined K loop: preload kt=0, then overlap kt+1 loads
        v16bf a_cur = *(const v16bf*)(pHr + aoff);
        v16bf b_cur[4];
        #pragma unroll
        for (int g = 0; g < 4; ++g)
            b_cur[g] = *(const v16bf*)(pWh + boff[g]);

        #pragma unroll
        for (int kt = 0; kt < KT_H; ++kt) {
            v16bf a_nxt;
            v16bf b_nxt[4];
            if (kt + 1 < KT_H) {
                a_nxt = *(const v16bf*)(pHr + aoff + (size_t)(kt + 1) * 2048);
                #pragma unroll
                for (int g = 0; g < 4; ++g)
                    b_nxt[g] = *(const v16bf*)(pWh + boff[g] + (size_t)(kt + 1) * 65536);
            }
            #pragma unroll
            for (int g = 0; g < 4; ++g)
                acc[g] = __builtin_amdgcn_wmma_f32_16x16x32_bf16(
                    false, a_cur, false, b_cur[g], (short)0, acc[g], false, false);
            if (kt + 1 < KT_H) {
                a_cur = a_nxt;
                #pragma unroll
                for (int g = 0; g < 4; ++g)
                    b_cur[g] = b_nxt[g];
            }
        }

        // pointwise LSTM cell update (wave-exclusive C), write new H into the
        // OTHER packed buffer -- no read/write hazard, one barrier suffices.
        #pragma unroll
        for (int v = 0; v < 8; ++v) {
            int m = mb + v;
            float I  = sigm(acc[0][v]);
            float F  = sigm(acc[1][v]);
            float O  = sigm(acc[2][v]);
            float Ct = tanhf(acc[3][v]);
            float c  = F * Cws[m * HID + h_col] + I * Ct;
            Cws[m * HID + h_col] = c;
            float hnew = O * tanhf(c);
            int sl = s_khb + (m & 15);
            pHw[(((size_t)s_kt * 4 + mt) * 32 + sl) * 16 + s_e] = (__bf16)hnew;
            if (t == SEQ - 1) Hf32[m * HID + h_col] = hnew;
        }

        grid_sync(bar, (unsigned)(R_BLOCKS * (t + 1)));   // writes visible
    }
}

// ---------------------------------------------------------------------------
// Kernel F: out[b][n] = H @ W_hq + b_q. Memory-bound (103 MB of W_hq read
// once, coalesced). H tiled through LDS; per-thread 64 fp32 accumulators.
// ---------------------------------------------------------------------------
__global__ __launch_bounds__(256) void out_proj(
    const float* __restrict__ Hf32, const float* __restrict__ Whq,
    const float* __restrict__ bq, float* __restrict__ out)
{
    __shared__ float lds[BATCH * 64];
    const int n = blockIdx.x * 256 + threadIdx.x;
    float acc[BATCH];
    #pragma unroll
    for (int b = 0; b < BATCH; ++b) acc[b] = 0.0f;

    for (int kc = 0; kc < HID / 64; ++kc) {
        __syncthreads();
        for (int i = threadIdx.x; i < BATCH * 64; i += 256) {
            int bb = i >> 6, kk = i & 63;
            lds[i] = Hf32[bb * HID + kc * 64 + kk];
        }
        __syncthreads();
        if (n < N_CLASS) {
            for (int kk = 0; kk < 64; ++kk) {
                float w = Whq[(size_t)(kc * 64 + kk) * N_CLASS + n];
                #pragma unroll
                for (int b = 0; b < BATCH; ++b)
                    acc[b] = fmaf(lds[b * 64 + kk], w, acc[b]);
            }
        }
    }
    if (n < N_CLASS) {
        float bias = bq[n];
        #pragma unroll
        for (int b = 0; b < BATCH; ++b)
            out[(size_t)b * N_CLASS + n] = acc[b] + bias;
    }
}

// ---------------------------------------------------------------------------
extern "C" void kernel_launch(void* const* d_in, const int* in_sizes, int n_in,
                              void* d_out, int out_size, void* d_ws, size_t ws_size,
                              hipStream_t stream)
{
    const int*   X   = (const int*)  d_in[0];
    const float* H   = (const float*)d_in[1];
    const float* C   = (const float*)d_in[2];
    const float* E   = (const float*)d_in[3];
    const float* Wxi = (const float*)d_in[4];
    const float* Wxf = (const float*)d_in[5];
    const float* Wxo = (const float*)d_in[6];
    const float* Wxc = (const float*)d_in[7];
    const float* Whi = (const float*)d_in[8];
    const float* Whf = (const float*)d_in[9];
    const float* Who = (const float*)d_in[10];
    const float* Whc = (const float*)d_in[11];
    const float* bi  = (const float*)d_in[12];
    const float* bfv = (const float*)d_in[13];
    const float* bo  = (const float*)d_in[14];
    const float* bc  = (const float*)d_in[15];
    const float* Whq = (const float*)d_in[16];
    const float* bq  = (const float*)d_in[17];
    float* out = (float*)d_out;

    char* ws = (char*)d_ws;
    size_t off = 0;
    auto alloc = [&](size_t bytes) -> void* {
        void* p = ws + off;
        off += (bytes + 255) & ~(size_t)255;
        return p;
    };
    __bf16*   pWx   = (__bf16*)  alloc((size_t)KT_X * NT_ALL * 32 * 16 * 2); //  1 MB
    __bf16*   pWh   = (__bf16*)  alloc((size_t)KT_H * NT_ALL * 32 * 16 * 2); //  2 MB
    __bf16*   pH0   = (__bf16*)  alloc((size_t)KT_H * 4 * 32 * 16 * 2);      // 64 KB
    __bf16*   pH1   = (__bf16*)  alloc((size_t)KT_H * 4 * 32 * 16 * 2);      // 64 KB
    float*    Cws   = (float*)   alloc((size_t)BATCH * HID * 4);
    float*    Hf    = (float*)   alloc((size_t)BATCH * HID * 4);
    unsigned* bar   = (unsigned*)alloc(256);
    float*    xproj = (float*)   alloc((size_t)SEQ * BATCH * NG * 4);        // 128 MB

    pack_weights<<<(KT_X + KT_H) * NT_ALL * 32 / 256, 256, 0, stream>>>(
        Wxi, Wxf, Wxo, Wxc, Whi, Whf, Who, Whc, pWx, pWh);
    init_state<<<(BATCH * HID) / 256, 256, 0, stream>>>(H, C, pH0, Cws, bar);
    xproj_kernel<<<(BATCH * SEQ / 16) * 4, 256, 0, stream>>>(
        X, E, pWx, bi, bfv, bo, bc, xproj);
    lstm_recurrence<<<R_BLOCKS, 256, 0, stream>>>(
        xproj, pWh, pH0, pH1, Cws, Hf, bar);
    out_proj<<<(N_CLASS + 255) / 256, 256, 0, stream>>>(Hf, Whq, bq, out);
}
